// FrequencySparseAttention_24567212933787
// MI455X (gfx1250) — matrix operand order repaired and checked by reading, not compiled, not run on hardware
//
#include <hip/hip_runtime.h>
#include <hip/hip_bf16.h>

// ---------------------------------------------------------------------------
// FrequencySparseAttention on gfx1250 (CDNA5), wave32 + WMMA f32_16x16x32_f16
//   B=2, L=2048, D=512, H=8, Dh=64, threshold 0.1
// Pipeline:
//   1) cvt f32->f16 for q_x/k_x/v_x and Wq/Wk/Wv/Wo into workspace
//   2) key-mask bias[b][l] = (mean|energy| > 0.1) ? 0 : -1e9
//   3) WMMA GEMM projections (ping-pong double-buffered fragment loads):
//      Q(=x@Wq.T+bq, pre-scaled 1/8), K, V(->Vt transposed per head)
//   4) flash attention (WMMA QK^T + P*V, online softmax, LDS P-transpose,
//      ping-pong K fragment buffers, early V fragment issue, K prefetch)
//   5) WMMA GEMM output projection -> d_out (f32)
// ---------------------------------------------------------------------------

#define Bb 2
#define Ll 2048
#define Dd 512
#define Hh 8
#define DHc 64

typedef __attribute__((ext_vector_type(16))) _Float16 v16h;
typedef __attribute__((ext_vector_type(8)))  _Float16 v8h;
typedef __attribute__((ext_vector_type(8)))  float    v8f;

union V16U { v16h v; v8h h[2]; };

__device__ __forceinline__ v8f vzero8() {
  v8f v;
#pragma unroll
  for (int i = 0; i < 8; ++i) v[i] = 0.0f;
  return v;
}

// A-fragment (16x32 f16, wave32). Lane row = lane&15, half = lane>>4.
// Elements 0-15 hold K = half*16 .. +15 (contiguous).
// rowp points at &A[row][k0].
__device__ __forceinline__ v16h frag_a(const _Float16* rowp, int hf) {
  V16U u;
  u.h[0] = *(const v8h*)(rowp + hf * 16);
  u.h[1] = *(const v8h*)(rowp + hf * 16 + 8);
  return u.v;
}

// B-fragment (32x16 f16, wave32). Lane col = lane&15 holds K = (lane>>4)*16..+15,
// contiguous in memory when column col of B is a contiguous row of the source.
__device__ __forceinline__ v16h frag_b(const _Float16* p) {
  V16U u;
  u.h[0] = *(const v8h*)(p);
  u.h[1] = *(const v8h*)(p + 8);
  return u.v;
}

__device__ __forceinline__ v8f wmma_f16(v16h a, v16h b, v8f c) {
  return __builtin_amdgcn_wmma_f32_16x16x32_f16(false, a, false, b, (short)0, c,
                                                false, false);
}

// ------------------------------ elementwise --------------------------------

__global__ void cvt_f32_f16(const float* __restrict__ src,
                            _Float16* __restrict__ dst, int n) {
  int i = blockIdx.x * blockDim.x + threadIdx.x;
  if (i < n) dst[i] = (_Float16)src[i];
}

__global__ void mask_kernel(const float* __restrict__ ec,
                            float* __restrict__ keybias, int n) {
  int i = blockIdx.x * blockDim.x + threadIdx.x;
  if (i < n) {
    const float4* e4 = (const float4*)(ec + (size_t)i * 16);
    float s = 0.0f;
#pragma unroll
    for (int j = 0; j < 4; ++j) {
      float4 v = e4[j];
      s += fabsf(v.x) + fabsf(v.y) + fabsf(v.z) + fabsf(v.w);
    }
    s *= (1.0f / 16.0f);
    // forward value of the straight-through mask: additive key bias
    keybias[i] = (s > 0.1f) ? 0.0f : -1e9f;
  }
}

// ------------------------------ WMMA GEMM ----------------------------------
// out[m][n] = (sum_k A[m][k] * W[n][k] + bias[n]) * scale,  K = Dd = 512
// mode 0: f16 row-major [M][512]      (Q, K)
// mode 1: f16 V-transposed per head:  Vt[b][h][dh][L]
// mode 2: f32 row-major [M][512]      (final output)
// Ping-pong double buffering: buffer A is reloaded while buffer B's WMMAs run
// and vice versa -> no register rotation copies, loads batched in clauses.
__global__ void __launch_bounds__(32)
wmma_proj(const _Float16* __restrict__ A, const _Float16* __restrict__ W,
          const float* __restrict__ bias, void* __restrict__ dst,
          int mode, float scale) {
  const int lane = threadIdx.x & 31;
  const int col = lane & 15, hf = lane >> 4;
  const int m0 = blockIdx.x * 16;
  const int n0 = blockIdx.y * 64;

  v8f acc[4] = {vzero8(), vzero8(), vzero8(), vzero8()};
  const _Float16* arow = A + (size_t)(m0 + col) * Dd;
  const _Float16* wrow[4];
#pragma unroll
  for (int t = 0; t < 4; ++t)
    wrow[t] = W + (size_t)(n0 + t * 16 + col) * Dd + hf * 16;

  // prologue: buffer A <- k0 = 0
  v16h afA = frag_a(arow, hf);
  v16h bfA[4];
#pragma unroll
  for (int t = 0; t < 4; ++t) bfA[t] = frag_b(wrow[t]);

  for (int k0 = 0; k0 < Dd - 64; k0 += 64) {
    v16h afB = frag_a(arow + k0 + 32, hf);       // buffer B <- k0+32
    v16h bfB[4];
#pragma unroll
    for (int t = 0; t < 4; ++t) bfB[t] = frag_b(wrow[t] + k0 + 32);
#pragma unroll
    for (int t = 0; t < 4; ++t) acc[t] = wmma_f16(afA, bfA[t], acc[t]);
    afA = frag_a(arow + k0 + 64, hf);            // buffer A <- k0+64
#pragma unroll
    for (int t = 0; t < 4; ++t) bfA[t] = frag_b(wrow[t] + k0 + 64);
#pragma unroll
    for (int t = 0; t < 4; ++t) acc[t] = wmma_f16(afB, bfB[t], acc[t]);
  }
  // epilogue: last two 32-chunks (k = Dd-64 in A, Dd-32 in B)
  {
    v16h afB = frag_a(arow + Dd - 32, hf);
    v16h bfB[4];
#pragma unroll
    for (int t = 0; t < 4; ++t) bfB[t] = frag_b(wrow[t] + Dd - 32);
#pragma unroll
    for (int t = 0; t < 4; ++t) acc[t] = wmma_f16(afA, bfA[t], acc[t]);
#pragma unroll
    for (int t = 0; t < 4; ++t) acc[t] = wmma_f16(afB, bfB[t], acc[t]);
  }

#pragma unroll
  for (int t = 0; t < 4; ++t) {
#pragma unroll
    for (int i = 0; i < 8; ++i) {
      int m = m0 + i + hf * 8;
      int n = n0 + t * 16 + col;
      float v = (acc[t][i] + bias[n]) * scale;
      if (mode == 0) {
        ((_Float16*)dst)[(size_t)m * Dd + n] = (_Float16)v;
      } else if (mode == 2) {
        ((float*)dst)[(size_t)m * Dd + n] = v;
      } else {  // Vt[b][h][dh][l]
        int bidx = m >> 11, l = m & (Ll - 1);
        int h = n >> 6, dh = n & (DHc - 1);
        ((_Float16*)dst)[(((size_t)bidx * Hh + h) * DHc + dh) * Ll + l] =
            (_Float16)v;
      }
    }
  }
}

// ----------------------------- flash attention -----------------------------

__device__ __forceinline__ void load_bk(v16h bk[2][2],
                                        const _Float16* kbase, int kt,
                                        int col) {
#pragma unroll
  for (int nt = 0; nt < 2; ++nt) {
    const _Float16* krow = kbase + (size_t)(kt + nt * 16 + col) * Dd;
    bk[nt][0] = frag_b(krow);
    bk[nt][1] = frag_b(krow + 32);
  }
}

// Process one 32-key tile: S = Q*K^T (+bias), online softmax update,
// P -> LDS transpose, O += P*V. Contains one uniform __syncthreads().
__device__ __forceinline__ void attn_step(
    v8f acc[4], float mr[8], float lr[8], v16h aq0, v16h aq1,
    const v16h bk[2][2], const _Float16* vbase, const float* kbb,
    _Float16* pl, int kt, int col, int hf) {
  // ---- scores: two 16x16 C tiles ------------------------------------------
  v8f s[2];
#pragma unroll
  for (int nt = 0; nt < 2; ++nt) {
    v8f c = vzero8();
    c = wmma_f16(aq0, bk[nt][0], c);   // dh chunk 0..31
    c = wmma_f16(aq1, bk[nt][1], c);   // dh chunk 32..63
    float bias = kbb[kt + nt * 16 + col];  // key = column = lane&15
#pragma unroll
    for (int i = 0; i < 8; ++i) c[i] += bias;
    s[nt] = c;
  }

  // ---- issue V fragments early (latency hidden by softmax) ----------------
  v16h bv[4];
#pragma unroll
  for (int t = 0; t < 4; ++t)
    bv[t] = frag_b(vbase + (size_t)(t * 16 + col) * Ll + kt + hf * 16);

  // ---- online softmax (rows live across the 16 lanes of each half) --------
  float tm[8], p0[8], p1[8], rs[8];
#pragma unroll
  for (int i = 0; i < 8; ++i) tm[i] = fmaxf(s[0][i], s[1][i]);
#pragma unroll
  for (int d = 1; d < 16; d <<= 1)
#pragma unroll
    for (int i = 0; i < 8; ++i)
      tm[i] = fmaxf(tm[i], __shfl_xor(tm[i], d, 16));

#pragma unroll
  for (int i = 0; i < 8; ++i) {
    float nm = fmaxf(mr[i], tm[i]);
    float corr = __expf(mr[i] - nm);
    p0[i] = __expf(s[0][i] - nm);
    p1[i] = __expf(s[1][i] - nm);
    rs[i] = p0[i] + p1[i];
    lr[i] *= corr;
    mr[i] = nm;
#pragma unroll
    for (int t = 0; t < 4; ++t) acc[t][i] *= corr;
  }
#pragma unroll
  for (int d = 1; d < 16; d <<= 1)
#pragma unroll
    for (int i = 0; i < 8; ++i) rs[i] += __shfl_xor(rs[i], d, 16);
#pragma unroll
  for (int i = 0; i < 8; ++i) lr[i] += rs[i];

  // ---- C-layout P -> A-layout via wave-private LDS tile --------------------
#pragma unroll
  for (int i = 0; i < 8; ++i) {
    pl[(i + hf * 8) * 32 + col] = (_Float16)p0[i];
    pl[(i + hf * 8) * 32 + col + 16] = (_Float16)p1[i];
  }
  __syncthreads();  // uniform across the block (same trip count for all waves)
  v16h ap = frag_a(pl + col * 32, hf);

  // ---- O += P * V ----------------------------------------------------------
#pragma unroll
  for (int t = 0; t < 4; ++t) acc[t] = wmma_f16(ap, bv[t], acc[t]);
}

// One wave per 16-query tile of one (b,h). 4 waves / block.
// Q is pre-scaled by 1/sqrt(Dh). keybias is the additive mask per key.
__global__ void __launch_bounds__(128)
attn_kernel(const _Float16* __restrict__ Qm, const _Float16* __restrict__ Km,
            const _Float16* __restrict__ Vt, const float* __restrict__ kb,
            _Float16* __restrict__ Om) {
  __shared__ _Float16 plds[4][16 * 32];  // wave-private P tiles (16 q x 32 k)

  const int wid = threadIdx.x >> 5;
  const int lane = threadIdx.x & 31;
  const int col = lane & 15, hf = lane >> 4;

  const int tile = blockIdx.x * 4 + wid;   // 2048 tiles total
  const int qt = tile & 127;               // L/16 = 128
  const int h = (tile >> 7) & 7;
  const int b = tile >> 10;

  // Q A-fragments: invariant over the key loop (K-dim = Dh = 64 -> 2 chunks)
  const _Float16* qrow =
      Qm + ((size_t)(b * Ll) + qt * 16 + col) * Dd + h * DHc;
  v16h aq0 = frag_a(qrow, hf);
  v16h aq1 = frag_a(qrow + 32, hf);

  const float* kbb = kb + b * Ll;
  const _Float16* kbase = Km + (size_t)(b * Ll) * Dd + h * DHc + hf * 16;
  const _Float16* vbase = Vt + ((size_t)b * Hh + h) * DHc * (size_t)Ll;
  _Float16* pl = plds[wid];

  float mr[8], lr[8];
#pragma unroll
  for (int i = 0; i < 8; ++i) { mr[i] = -1e30f; lr[i] = 0.0f; }
  v8f acc[4] = {vzero8(), vzero8(), vzero8(), vzero8()};

  // ping-pong K fragment buffers: A holds tile kt, B holds tile kt+32
  v16h bkA[2][2], bkB[2][2];
  load_bk(bkA, kbase, 0, col);

  for (int kt = 0; kt < Ll - 64; kt += 64) {
    load_bk(bkB, kbase, kt + 32, col);
    if (kt + 128 < Ll)  // warm L2 two iterations ahead (global_prefetch_b8)
      __builtin_prefetch(kbase + (size_t)(kt + 128 + col) * Dd, 0, 3);
    attn_step(acc, mr, lr, aq0, aq1, bkA, vbase, kbb, pl, kt, col, hf);
    load_bk(bkA, kbase, kt + 64, col);
    attn_step(acc, mr, lr, aq0, aq1, bkB, vbase, kbb, pl, kt + 32, col, hf);
  }
  // epilogue: last two tiles
  load_bk(bkB, kbase, Ll - 32, col);
  attn_step(acc, mr, lr, aq0, aq1, bkA, vbase, kbb, pl, Ll - 64, col, hf);
  attn_step(acc, mr, lr, aq0, aq1, bkB, vbase, kbb, pl, Ll - 32, col, hf);

  // ---- normalize + merge heads -------------------------------------------
#pragma unroll
  for (int i = 0; i < 8; ++i) {
    float inv = 1.0f / lr[i];
#pragma unroll
    for (int t = 0; t < 4; ++t) {
      Om[((size_t)(b * Ll) + qt * 16 + i + hf * 8) * Dd + h * DHc + t * 16 +
         col] = (_Float16)(acc[t][i] * inv);
    }
  }
}

// ------------------------------- launcher ----------------------------------

extern "C" void kernel_launch(void* const* d_in, const int* in_sizes, int n_in,
                              void* d_out, int out_size, void* d_ws,
                              size_t ws_size, hipStream_t stream) {
  (void)in_sizes; (void)n_in; (void)out_size; (void)ws_size;
  const float* q_x = (const float*)d_in[0];
  const float* k_x = (const float*)d_in[1];
  const float* v_x = (const float*)d_in[2];
  const float* ec  = (const float*)d_in[3];
  const float* Wq  = (const float*)d_in[4];
  const float* bq  = (const float*)d_in[5];
  const float* Wk  = (const float*)d_in[6];
  const float* bk  = (const float*)d_in[7];
  const float* Wv  = (const float*)d_in[8];
  const float* bv  = (const float*)d_in[9];
  const float* Wo  = (const float*)d_in[10];
  const float* bo  = (const float*)d_in[11];

  const size_t NX = (size_t)Bb * Ll * Dd;  // 2,097,152
  const size_t NW = (size_t)Dd * Dd;       // 262,144

  char* w = (char*)d_ws;
  size_t off = 0;
  auto take = [&](size_t elems, size_t elsz) -> void* {
    void* p = w + off;
    off += ((elems * elsz + 255) & ~(size_t)255);
    return p;
  };
  _Float16* Xq   = (_Float16*)take(NX, 2);
  _Float16* Xk   = (_Float16*)take(NX, 2);
  _Float16* Xv   = (_Float16*)take(NX, 2);
  _Float16* Wq16 = (_Float16*)take(NW, 2);
  _Float16* Wk16 = (_Float16*)take(NW, 2);
  _Float16* Wv16 = (_Float16*)take(NW, 2);
  _Float16* Wo16 = (_Float16*)take(NW, 2);
  _Float16* Qm   = (_Float16*)take(NX, 2);
  _Float16* Kmm  = (_Float16*)take(NX, 2);
  _Float16* Vt   = (_Float16*)take(NX, 2);
  _Float16* Om   = (_Float16*)take(NX, 2);
  float*    kbias = (float*)take((size_t)Bb * Ll, 4);
  // total ~30.3 MB of workspace

  const dim3 cb(256);
  cvt_f32_f16<<<dim3((unsigned)((NX + 255) / 256)), cb, 0, stream>>>(q_x, Xq, (int)NX);
  cvt_f32_f16<<<dim3((unsigned)((NX + 255) / 256)), cb, 0, stream>>>(k_x, Xk, (int)NX);
  cvt_f32_f16<<<dim3((unsigned)((NX + 255) / 256)), cb, 0, stream>>>(v_x, Xv, (int)NX);
  cvt_f32_f16<<<dim3((unsigned)((NW + 255) / 256)), cb, 0, stream>>>(Wq, Wq16, (int)NW);
  cvt_f32_f16<<<dim3((unsigned)((NW + 255) / 256)), cb, 0, stream>>>(Wk, Wk16, (int)NW);
  cvt_f32_f16<<<dim3((unsigned)((NW + 255) / 256)), cb, 0, stream>>>(Wv, Wv16, (int)NW);
  cvt_f32_f16<<<dim3((unsigned)((NW + 255) / 256)), cb, 0, stream>>>(Wo, Wo16, (int)NW);
  mask_kernel<<<dim3((Bb * Ll + 255) / 256), cb, 0, stream>>>(ec, kbias, Bb * Ll);

  const dim3 gg((Bb * Ll) / 16, Dd / 64);  // (256, 8)
  const dim3 gw(32);
  // Q scaled by 1/sqrt(Dh) = 0.125 (bias folded in correctly: (xW^T+b)/8)
  wmma_proj<<<gg, gw, 0, stream>>>(Xq, Wq16, bq, (void*)Qm, 0, 0.125f);
  wmma_proj<<<gg, gw, 0, stream>>>(Xk, Wk16, bk, (void*)Kmm, 0, 1.0f);
  wmma_proj<<<gg, gw, 0, stream>>>(Xv, Wv16, bv, (void*)Vt, 1, 1.0f);

  attn_kernel<<<dim3((Bb * Hh * (Ll / 16)) / 4), dim3(128), 0, stream>>>(
      Qm, Kmm, Vt, kbias, Om);

  wmma_proj<<<gg, gw, 0, stream>>>(Om, Wo16, bo, d_out, 2, 1.0f);
}
